// DGCNN_36575941492861
// MI455X (gfx1250) — compile-verified, matrix operand order checked by reading, not run on hardware
//
#include <hip/hip_runtime.h>
#include <hip/hip_bf16.h>
#include <math.h>

// ---------------------------------------------------------------------------
// DGCNN forward for MI455X (gfx1250), v3.
// - All GEMM work on v_wmma_f32_16x16x32_f16, fragments via aligned b128 loads
//   from pre-converted f16 buffers (ISA 7.12.2 layout), weights pre-transposed.
// - Edge conv processes NT=4 output tiles per wave with VGPR-resident A frags.
// - KNN top-K selection uses both half-waves + LDS two-pointer merge.
// ---------------------------------------------------------------------------

typedef __attribute__((ext_vector_type(16))) _Float16 v16h;
typedef __attribute__((ext_vector_type(8)))  _Float16 v8h;
typedef __attribute__((ext_vector_type(8)))  float    v8f;

union HF { v16h v; v8h h[2]; };

#define BB   16
#define PP   2048
#define KNN  20
#define TOPK 20

// Fragment element -> K index within a 32-wide K chunk (ISA 7.12.2):
// per lane: two contiguous runs of 8 halves, [kb,kb+8) and [kb+16,kb+24),
// kb = (lane>=16)*8.
__device__ __forceinline__ int fk(int lane, int e) {
  return ((lane >> 4) << 3) + e + ((e >= 8) ? 8 : 0);
}

// ---------------- f32 -> f16 with zero padding to dp ----------------
__global__ void dgcnn_tohalf(const float* __restrict__ X, _Float16* __restrict__ Xh,
                             int d, int dp, int nrows) {
  int i = blockIdx.x * blockDim.x + threadIdx.x;
  if (i >= nrows * dp) return;
  int r = i / dp, k = i - r * dp;
  Xh[i] = (_Float16)((k < d) ? X[(size_t)r * d + k] : 0.f);
}

// ---------------- W[K2][N] f32 -> Wt[N][K2P] f16 (transpose+pad) -----
__global__ void dgcnn_transw(const float* __restrict__ W, _Float16* __restrict__ Wt,
                             int K2, int K2P, int N) {
  int i = blockIdx.x * blockDim.x + threadIdx.x;
  if (i >= N * K2P) return;
  int n = i / K2P, k = i - n * K2P;
  Wt[i] = (_Float16)((k < K2) ? W[(size_t)k * N + n] : 0.f);
}

// ---------------- squared norms (distance ranking) -------------------
__global__ void dgcnn_sumsq(const float* __restrict__ X, float* __restrict__ sq, int d) {
  int i = blockIdx.x * blockDim.x + threadIdx.x;
  if (i >= BB * PP) return;
  const float* p = X + (size_t)i * d;
  float s = 0.f;
  for (int k = 0; k < d; ++k) s += p[k] * p[k];
  sq[i] = s;
}

// ---------------- fused Gram (WMMA) + dual-half top-K selection ------
template <int DP>  // padded feature dim, multiple of 32
__global__ void __launch_bounds__(128) dgcnn_knn_wmma(const _Float16* __restrict__ Xh,
                                                      const float* __restrict__ sq,
                                                      int* __restrict__ out_idx) {
  constexpr int NCH = DP >> 5;
  __shared__ float tile[4][16 * 16];
  __shared__ float sdist[4][2][16][TOPK];
  __shared__ int   sidx [4][2][16][TOPK];
  const int wave = threadIdx.x >> 5;
  const int lane = threadIdx.x & 31;
  const int m = lane & 15, hi = lane >> 4;
  const int kb = hi << 3;
  const int t    = blockIdx.x * 4 + wave;        // 16-row query tile
  const int b    = t / (PP / 16);
  const int row0 = (t % (PP / 16)) * 16;

  // A fragments (query rows): 2 x b128 per chunk, VGPR-resident for the sweep.
  HF afrag[NCH];
  {
    const _Float16* ar = Xh + (size_t)(b * PP + row0 + m) * DP;
#pragma unroll
    for (int c = 0; c < NCH; ++c) {
      afrag[c].h[0] = *(const v8h*)(ar + c * 32 + kb);
      afrag[c].h[1] = *(const v8h*)(ar + c * 32 + kb + 16);
    }
  }

  // Both half-waves keep a sorted top-K of their half of the candidates:
  // lane m scans tile cols [0,8), lane m+16 scans cols [8,16).
  float bd[TOPK]; int bi[TOPK];
#pragma unroll
  for (int s = 0; s < TOPK; ++s) { bd[s] = 3.0e30f; bi[s] = 0; }

  for (int ct = 0; ct < PP / 16; ++ct) {
    const int col0 = ct * 16;
    const _Float16* cr = Xh + (size_t)(b * PP + col0 + m) * DP;
    if (ct + 1 < PP / 16) __builtin_prefetch(cr + 16 * DP, 0, 3);  // global_prefetch_b8

    v8f acc = {};
#pragma unroll
    for (int c = 0; c < NCH; ++c) {
      HF bf;
      bf.h[0] = *(const v8h*)(cr + c * 32 + kb);
      bf.h[1] = *(const v8h*)(cr + c * 32 + kb + 16);
      acc = __builtin_amdgcn_wmma_f32_16x16x32_f16(false, afrag[c].v, false, bf.v,
                                                   (short)0, acc, false, false);
    }

    // rank key: d2[col] - 2*dot (d2[row] constant per row -> rank-invariant)
    const float dc = sq[b * PP + col0 + m];
#pragma unroll
    for (int v = 0; v < 8; ++v) {
      int row = row0 + v + kb;                    // C layout: M = v + 8*hi, N = m
      float dv = dc - 2.f * acc[v];
      if (row == col0 + m) dv = 3.0e30f;          // exclude self
      tile[wave][(v + kb) * 16 + m] = dv;         // tile[M][N], transposed via LDS
    }
    __syncthreads();
#pragma unroll
    for (int c = 0; c < 8; ++c) {                 // all 32 lanes select
      float dv = tile[wave][m * 16 + kb + c];
      if (dv < bd[TOPK - 1]) {                    // register-resident sorted insert
        float cv = dv; int ci = col0 + kb + c;
#pragma unroll
        for (int s = 0; s < TOPK; ++s) {
          if (cv < bd[s]) {
            float td = bd[s]; int ti = bi[s];
            bd[s] = cv; bi[s] = ci; cv = td; ci = ti;
          }
        }
      }
    }
    __syncthreads();
  }

  // merge the two sorted per-half lists (two-pointer, 20 steps) via LDS
#pragma unroll
  for (int s = 0; s < TOPK; ++s) {
    sdist[wave][hi][m][s] = bd[s];
    sidx [wave][hi][m][s] = bi[s];
  }
  __syncthreads();
  if (lane < 16) {
    const float* da = sdist[wave][0][lane]; const int* ia = sidx[wave][0][lane];
    const float* db = sdist[wave][1][lane]; const int* ib = sidx[wave][1][lane];
    int* op = out_idx + (size_t)(b * PP + row0 + lane) * KNN;
    int pa = 0, pb = 0;
    for (int s = 0; s < TOPK; ++s) {
      float va = da[pa], vb = db[pb];
      if (va <= vb) { op[s] = ia[pa]; ++pa; }
      else          { op[s] = ib[pb]; ++pb; }
    }
  }
}

// ---------------- edge conv: msg=[xi, xj-xi] @ W, BN, ReLU, max over K ----
// NT output tiles per wave; A fragments stay VGPR-resident across tiles.
template <int CIN, int COUT, int NT>
__global__ void __launch_bounds__(32) dgcnn_edgeconv_wmma(
    const _Float16* __restrict__ Xh, const int* __restrict__ nbr,
    const _Float16* __restrict__ Wt,   // [COUT][K2P] f16 (transposed, padded)
    const float* __restrict__ bias,
    const float* __restrict__ bnb, const float* __restrict__ bng,
    const float* __restrict__ bnm, const float* __restrict__ bnv,
    float* __restrict__ Xout) {
  constexpr int CIN2 = 2 * CIN;
  constexpr int NCH  = (CIN2 + 31) >> 5;
  constexpr int K2P  = NCH * 32;
  constexpr int DP   = (CIN + 31) & ~31;          // Xh row stride
  __shared__ float hbuf[16 * 16];
  const int lane = threadIdx.x & 31;
  const int m = lane & 15, hi = lane >> 4;
  const int kb = hi << 3;
  const int t     = blockIdx.x;       // 16-row tile of (point, neighbor) pairs
  const int nbase = blockIdx.y * 16 * NT;

  const int r    = t * 16 + m;
  const int plin = r / KNN;
  const int kk   = r - plin * KNN;
  const int b    = plin / PP;
  const int j    = nbr[(size_t)plin * KNN + kk];
  const _Float16* xi = Xh + (size_t)plin * DP;
  const _Float16* xj = Xh + (size_t)(b * PP + j) * DP;

  // Build message A-fragments once; reuse across NT output tiles.
  HF afrag[NCH];
#pragma unroll
  for (int c = 0; c < NCH; ++c) {
    if constexpr (CIN % 32 == 0) {
      const int k0 = c * 32;                       // compile-time after unroll
      if (k0 + 32 <= CIN) {                        // pure xi chunk
        afrag[c].h[0] = *(const v8h*)(xi + k0 + kb);
        afrag[c].h[1] = *(const v8h*)(xi + k0 + kb + 16);
      } else {                                     // pure (xj - xi) chunk
        const int kd = k0 - CIN;
        v8h i0 = *(const v8h*)(xi + kd + kb);
        v8h i1 = *(const v8h*)(xi + kd + kb + 16);
        v8h j0 = *(const v8h*)(xj + kd + kb);
        v8h j1 = *(const v8h*)(xj + kd + kb + 16);
        afrag[c].h[0] = j0 - i0;                   // v_pk_add_f16 (neg)
        afrag[c].h[1] = j1 - i1;
      }
    } else {                                       // conv1 (cin=3): elementwise
#pragma unroll
      for (int e = 0; e < 16; ++e) {
        int k = (c << 5) + fk(lane, e);
        _Float16 av = (_Float16)0.f;
        if (k < CIN)       av = xi[k];
        else if (k < CIN2) av = xj[k - CIN] - xi[k - CIN];
        afrag[c].v[e] = av;
      }
    }
  }

  for (int it = 0; it < NT; ++it) {
    const int n0 = nbase + it * 16;
    const _Float16* wr = Wt + (size_t)(n0 + m) * K2P;
    v8f acc = {};
#pragma unroll
    for (int c = 0; c < NCH; ++c) {
      HF bf;
      bf.h[0] = *(const v8h*)(wr + c * 32 + kb);
      bf.h[1] = *(const v8h*)(wr + c * 32 + kb + 16);
      acc = __builtin_amdgcn_wmma_f32_16x16x32_f16(false, afrag[c].v, false, bf.v,
                                                   (short)0, acc, false, false);
    }

    const int n = n0 + m;
    const float sc  = rsqrtf(bnv[n] + 1e-5f) * bng[n];
    const float sh  = bnb[n] - bnm[n] * sc;
    const float bsv = bias[n];
#pragma unroll
    for (int v = 0; v < 8; ++v) {
      float h = (acc[v] + bsv) * sc + sh;
      hbuf[(v + kb) * 16 + m] = fmaxf(h, 0.f);
    }
    __syncthreads();
    // max over K: rows in this tile span <=2 points; ReLU>=0 so int-bit
    // atomicMax on zero-initialized output is an exact float max.
    if (lane < 16) {
      int curp = (t * 16) / KNN;
      float cm = 0.f;
#pragma unroll
      for (int rt = 0; rt < 16; ++rt) {
        int p2 = (t * 16 + rt) / KNN;
        float hv = hbuf[rt * 16 + lane];
        if (p2 != curp) {
          atomicMax((int*)&Xout[(size_t)curp * COUT + n0 + lane], __float_as_int(cm));
          curp = p2; cm = 0.f;
        }
        cm = fmaxf(cm, hv);
      }
      atomicMax((int*)&Xout[(size_t)curp * COUT + n0 + lane], __float_as_int(cm));
    }
    __syncthreads();
  }
}

// ---------------- global max pool over P into 512-d descriptor ----------
__global__ void dgcnn_pool(const float* __restrict__ x1, const float* __restrict__ x2,
                           const float* __restrict__ x3, const float* __restrict__ x4,
                           float* __restrict__ g) {
  int b = blockIdx.x >> 1;
  int c = ((blockIdx.x & 1) << 8) + threadIdx.x;   // 0..511
  float mx = -3.0e30f;
  for (int p = 0; p < PP; ++p) {
    size_t bp = (size_t)(b * PP + p);
    float v;
    if (c < 64)       v = x1[bp * 64 + c];
    else if (c < 128) v = x2[bp * 64 + (c - 64)];
    else if (c < 256) v = x3[bp * 128 + (c - 128)];
    else              v = x4[bp * 256 + (c - 256)];
    mx = fmaxf(mx, v);
  }
  g[b * 512 + c] = mx;
}

// ---------------- head linear layers: M=16 == one WMMA tile -------------
__global__ void __launch_bounds__(32) dgcnn_linear16_wmma(
    const _Float16* __restrict__ Ah,   // [16][Kd] f16
    const _Float16* __restrict__ Wt,   // [N][Kd]  f16 (transposed)
    const float* __restrict__ bias,
    const float* __restrict__ bnb, const float* __restrict__ bng,
    const float* __restrict__ bnm, const float* __restrict__ bnv,
    float* __restrict__ Out, int Kd, int N) {
  const int lane = threadIdx.x & 31;
  const int m = lane & 15, hi = lane >> 4;
  const int kb = hi << 3;
  const int n0 = blockIdx.x * 16;
  const _Float16* ar = Ah + (size_t)m * Kd;
  const _Float16* wr = Wt + (size_t)(n0 + m) * Kd;
  v8f acc = {};
  for (int c = 0; c < (Kd >> 5); ++c) {
    HF af, bf;
    af.h[0] = *(const v8h*)(ar + c * 32 + kb);
    af.h[1] = *(const v8h*)(ar + c * 32 + kb + 16);
    bf.h[0] = *(const v8h*)(wr + c * 32 + kb);
    bf.h[1] = *(const v8h*)(wr + c * 32 + kb + 16);
    acc = __builtin_amdgcn_wmma_f32_16x16x32_f16(false, af.v, false, bf.v,
                                                 (short)0, acc, false, false);
  }
  const int n = n0 + m;
  const float sc  = rsqrtf(bnv[n] + 1e-5f) * bng[n];
  const float sh  = bnb[n] - bnm[n] * sc;
  const float bsv = bias[n];
#pragma unroll
  for (int v = 0; v < 8; ++v) {
    float h = fmaxf((acc[v] + bsv) * sc + sh, 0.f);
    Out[(size_t)(v + kb) * N + n] = h;
  }
}

// ---------------- final 512->40 + log_softmax ---------------------------
__global__ void dgcnn_logits(const float* __restrict__ H, const float* __restrict__ W,
                             const float* __restrict__ bias, float* __restrict__ out) {
  __shared__ float sl[40];
  __shared__ float red[2];
  const int b = blockIdx.x;
  const int c = threadIdx.x;
  float v = 0.f;
  if (c < 40) {
    const float* h = H + (size_t)b * 512;
    float s = bias[c];
    for (int k = 0; k < 512; ++k) s += h[k] * W[(size_t)k * 40 + c];
    sl[c] = s; v = s;
  }
  __syncthreads();
  if (c == 0) {
    float mx = sl[0];
    for (int k = 1; k < 40; ++k) mx = fmaxf(mx, sl[k]);
    float se = 0.f;
    for (int k = 0; k < 40; ++k) se += expf(sl[k] - mx);
    red[0] = mx; red[1] = logf(se);
  }
  __syncthreads();
  if (c < 40) out[(size_t)b * 40 + c] = v - red[0] - red[1];
}

// ---------------------------------------------------------------------------
// Host launcher. Input leaf order assumes jax pytree flattening (dict keys
// sorted): batch, bn1{beta,gamma,mean,var}, bn2{...}, conv1{W,b,bn{...}},
// conv2{...}, conv3{...}, conv4{...}, fc1{W,b}, fc2{W,b}, lin1{W,b}, x.
// ---------------------------------------------------------------------------
extern "C" void kernel_launch(void* const* d_in, const int* in_sizes, int n_in,
                              void* d_out, int out_size, void* d_ws, size_t ws_size,
                              hipStream_t stream) {
  const float* bn1b = (const float*)d_in[1];
  const float* bn1g = (const float*)d_in[2];
  const float* bn1m = (const float*)d_in[3];
  const float* bn1v = (const float*)d_in[4];
  const float* bn2b = (const float*)d_in[5];
  const float* bn2g = (const float*)d_in[6];
  const float* bn2m = (const float*)d_in[7];
  const float* bn2v = (const float*)d_in[8];
  const float* c1W = (const float*)d_in[9];  const float* c1b = (const float*)d_in[10];
  const float* c1bnb = (const float*)d_in[11]; const float* c1bng = (const float*)d_in[12];
  const float* c1bnm = (const float*)d_in[13]; const float* c1bnv = (const float*)d_in[14];
  const float* c2W = (const float*)d_in[15]; const float* c2b = (const float*)d_in[16];
  const float* c2bnb = (const float*)d_in[17]; const float* c2bng = (const float*)d_in[18];
  const float* c2bnm = (const float*)d_in[19]; const float* c2bnv = (const float*)d_in[20];
  const float* c3W = (const float*)d_in[21]; const float* c3b = (const float*)d_in[22];
  const float* c3bnb = (const float*)d_in[23]; const float* c3bng = (const float*)d_in[24];
  const float* c3bnm = (const float*)d_in[25]; const float* c3bnv = (const float*)d_in[26];
  const float* c4W = (const float*)d_in[27]; const float* c4b = (const float*)d_in[28];
  const float* c4bnb = (const float*)d_in[29]; const float* c4bng = (const float*)d_in[30];
  const float* c4bnm = (const float*)d_in[31]; const float* c4bnv = (const float*)d_in[32];
  const float* fc1W = (const float*)d_in[33]; const float* fc1b = (const float*)d_in[34];
  const float* fc2W = (const float*)d_in[35]; const float* fc2b = (const float*)d_in[36];
  const float* lin1W = (const float*)d_in[37]; const float* lin1b = (const float*)d_in[38];
  const float* x = (const float*)d_in[39];

  float* ws = (float*)d_ws;
  float* x1 = ws;                                // [B*P,  64] f32
  float* x2 = x1 + (size_t)BB * PP * 64;         // [B*P,  64]
  float* x3 = x2 + (size_t)BB * PP * 64;         // [B*P, 128]
  float* x4 = x3 + (size_t)BB * PP * 128;        // [B*P, 256]
  float* sq = x4 + (size_t)BB * PP * 256;        // [B*P]
  float* g  = sq + BB * PP;                      // [16, 512]
  float* h1 = g + 16 * 512;                      // [16, 1024]
  float* h2 = h1 + 16 * 1024;                    // [16, 512]
  int* nbr  = (int*)(h2 + 16 * 512);             // [B*P, K] int32
  _Float16* Xh = (_Float16*)(nbr + (size_t)BB * PP * KNN);  // [B*P, <=128] f16
  _Float16* Wt = Xh + (size_t)BB * PP * 128;     // [N][K2P] f16, <= 512K halves
  _Float16* Ah = Wt + (size_t)512 * 1024;        // [16][<=1024] f16

  // zero-init atomic-max targets (graph-capturable memset node)
  hipMemsetAsync(x1, 0, (size_t)BB * PP * (64 + 64 + 128 + 256) * sizeof(float), stream);

  const int NP = BB * PP;                        // 32768 points
  const dim3 knnB(128), knnG(NP / 16 / 4);
  const dim3 ecB(32);
  const int rtiles = NP * KNN / 16;              // 40960

  // ---- stage 1: knn on 3-d coords, edgeconv 6->64 ----
  dgcnn_tohalf<<<(NP * 32 + 255) / 256, 256, 0, stream>>>(x, Xh, 3, 32, NP);
  dgcnn_sumsq<<<(NP + 255) / 256, 256, 0, stream>>>(x, sq, 3);
  dgcnn_knn_wmma<32><<<knnG, knnB, 0, stream>>>(Xh, sq, nbr);
  dgcnn_transw<<<(64 * 32 + 255) / 256, 256, 0, stream>>>(c1W, Wt, 6, 32, 64);
  dgcnn_edgeconv_wmma<3, 64, 4><<<dim3(rtiles, 1), ecB, 0, stream>>>(
      Xh, nbr, Wt, c1b, c1bnb, c1bng, c1bnm, c1bnv, x1);

  // ---- stage 2: knn on 64-d, edgeconv 128->64 ----
  dgcnn_tohalf<<<(NP * 64 + 255) / 256, 256, 0, stream>>>(x1, Xh, 64, 64, NP);
  dgcnn_sumsq<<<(NP + 255) / 256, 256, 0, stream>>>(x1, sq, 64);
  dgcnn_knn_wmma<64><<<knnG, knnB, 0, stream>>>(Xh, sq, nbr);
  dgcnn_transw<<<(64 * 128 + 255) / 256, 256, 0, stream>>>(c2W, Wt, 128, 128, 64);
  dgcnn_edgeconv_wmma<64, 64, 4><<<dim3(rtiles, 1), ecB, 0, stream>>>(
      Xh, nbr, Wt, c2b, c2bnb, c2bng, c2bnm, c2bnv, x2);

  // ---- stage 3: knn on 64-d, edgeconv 128->128 ----
  dgcnn_tohalf<<<(NP * 64 + 255) / 256, 256, 0, stream>>>(x2, Xh, 64, 64, NP);
  dgcnn_sumsq<<<(NP + 255) / 256, 256, 0, stream>>>(x2, sq, 64);
  dgcnn_knn_wmma<64><<<knnG, knnB, 0, stream>>>(Xh, sq, nbr);
  dgcnn_transw<<<(128 * 128 + 255) / 256, 256, 0, stream>>>(c3W, Wt, 128, 128, 128);
  dgcnn_edgeconv_wmma<64, 128, 4><<<dim3(rtiles, 2), ecB, 0, stream>>>(
      Xh, nbr, Wt, c3b, c3bnb, c3bng, c3bnm, c3bnv, x3);

  // ---- stage 4: knn on 128-d, edgeconv 256->256 ----
  dgcnn_tohalf<<<(NP * 128 + 255) / 256, 256, 0, stream>>>(x3, Xh, 128, 128, NP);
  dgcnn_sumsq<<<(NP + 255) / 256, 256, 0, stream>>>(x3, sq, 128);
  dgcnn_knn_wmma<128><<<knnG, knnB, 0, stream>>>(Xh, sq, nbr);
  dgcnn_transw<<<(256 * 256 + 255) / 256, 256, 0, stream>>>(c4W, Wt, 256, 256, 256);
  dgcnn_edgeconv_wmma<128, 256, 4><<<dim3(rtiles, 4), ecB, 0, stream>>>(
      Xh, nbr, Wt, c4b, c4bnb, c4bng, c4bnm, c4bnv, x4);

  // ---- global pooling + head MLP (M=16 WMMA tiles) ----
  dgcnn_pool<<<BB * 2, 256, 0, stream>>>(x1, x2, x3, x4, g);

  dgcnn_tohalf<<<(16 * 512 + 255) / 256, 256, 0, stream>>>(g, Ah, 512, 512, 16);
  dgcnn_transw<<<(1024 * 512 + 255) / 256, 256, 0, stream>>>(lin1W, Wt, 512, 512, 1024);
  dgcnn_linear16_wmma<<<1024 / 16, 32, 0, stream>>>(Ah, Wt, lin1b,
      bn1b, bn1g, bn1m, bn1v, h1, 512, 1024);

  dgcnn_tohalf<<<(16 * 1024 + 255) / 256, 256, 0, stream>>>(h1, Ah, 1024, 1024, 16);
  dgcnn_transw<<<(512 * 1024 + 255) / 256, 256, 0, stream>>>(fc1W, Wt, 1024, 1024, 512);
  dgcnn_linear16_wmma<<<512 / 16, 32, 0, stream>>>(Ah, Wt, fc1b,
      bn2b, bn2g, bn2m, bn2v, h2, 1024, 512);

  dgcnn_logits<<<BB, 64, 0, stream>>>(h2, fc2W, fc2b, (float*)d_out);
}
